// GraphProp_39290360824584
// MI455X (gfx1250) — compile-verified, matrix operand order checked by reading, not run on hardware
//
#include <hip/hip_runtime.h>
#include <stdint.h>

#define HD   128
#define HD2  256
#define HD3  384
#define ETILE 64
#define NTILE 16

typedef __attribute__((ext_vector_type(16))) __bf16 v16bf;
typedef __attribute__((ext_vector_type(8)))  __bf16 v8bf;
typedef __attribute__((ext_vector_type(4)))  __bf16 v4bf;
typedef __attribute__((ext_vector_type(8)))  float  v8f;
typedef __attribute__((ext_vector_type(4)))  float  v4f;

// Build a 16x16x32 bf16 WMMA operand fragment from two 16-byte chunks:
// per-lane K = {k0..k0+7} and {k0+16..k0+23} per the CDNA5 16-bit A/B layout
// (lanes 0-15 use K-base 0, lanes 16-31 use K-base 8 — handled by caller).
static __device__ __forceinline__ v16bf load_frag(const __bf16* p0, const __bf16* p1) {
  v8bf a = *(const v8bf*)p0;
  v8bf b = *(const v8bf*)p1;
  v16bf r;
#pragma unroll
  for (int i = 0; i < 8; ++i) { r[i] = a[i]; r[8 + i] = b[i]; }
  return r;
}

// f32 GEMM via bf16 hi/lo split: 3 bf16 WMMAs ~ one f32-precision MMA step.
static __device__ __forceinline__ v8f wmma_split(v16bf ah, v16bf al, v16bf bh, v16bf bl, v8f c) {
  c = __builtin_amdgcn_wmma_f32_16x16x32_bf16(false, ah, false, bh, (short)0, c, false, false);
  c = __builtin_amdgcn_wmma_f32_16x16x32_bf16(false, ah, false, bl, (short)0, c, false, false);
  c = __builtin_amdgcn_wmma_f32_16x16x32_bf16(false, al, false, bh, (short)0, c, false, false);
  return c;
}

// Pack 4 f32 into bf16 hi + residual lo nibbles (8-byte LDS stores).
static __device__ __forceinline__ void split4(v4f v, v4bf* ph, v4bf* pl) {
  v4bf h, l;
#pragma unroll
  for (int c = 0; c < 4; ++c) {
    float f = v[c];
    __bf16 hb = (__bf16)f;
    h[c] = hb;
    l[c] = (__bf16)(f - (float)hb);
  }
  *ph = h; *pl = l;
}

__global__ void zero_f32_kernel(float* p, long n) {
  long i = (long)blockIdx.x * blockDim.x + threadIdx.x;
  long s = (long)gridDim.x * blockDim.x;
  for (; i < n; i += s) p[i] = 0.0f;
}

__global__ void zero_i32_kernel(int* p, long n) {
  long i = (long)blockIdx.x * blockDim.x + threadIdx.x;
  long s = (long)gridDim.x * blockDim.x;
  for (; i < n; i += s) p[i] = 0;
}

__global__ void deg_kernel(const long long* __restrict__ edges, int* __restrict__ deg, long nE) {
  long i = (long)blockIdx.x * blockDim.x + threadIdx.x;
  if (i < nE) {
    int d = (int)edges[2 * i + 1];
    atomicAdd(&deg[d], 1);
  }
}

// Split f32 -> bf16 hi + bf16 lo (residual).
__global__ void cvt_kernel(const float* __restrict__ src, __bf16* __restrict__ hi,
                           __bf16* __restrict__ lo, long n) {
  long i = (long)blockIdx.x * blockDim.x + threadIdx.x;
  long s = (long)gridDim.x * blockDim.x;
  for (; i < n; i += s) {
    float f = src[i];
    __bf16 h = (__bf16)f;
    hi[i] = h;
    lo[i] = (__bf16)(f - (float)h);
  }
}

// ---- Edge message GEMM fused with segment-sum ----
// Block: 64 edges x 256 outputs, 8 waves. A = gathered [h_src||h_dst] (K=256),
// B = W_msg row-major (output o contiguous in K). atomicAdd fragments into agg.
__global__ __launch_bounds__(256, 2) void msg_kernel(
    const float* __restrict__ h, const long long* __restrict__ edges,
    const __bf16* __restrict__ Wh, const __bf16* __restrict__ Wl,
    float* __restrict__ agg, int nEdges)
{
  __shared__ __align__(16) __bf16 Ahi[ETILE][HD2 + 8];
  __shared__ __align__(16) __bf16 Alo[ETILE][HD2 + 8];
  __shared__ int dstLds[ETILE];

  const int tid  = threadIdx.x;
  const int base = blockIdx.x * ETILE;

  // Gather + bf16 split: 128 (edge,half-row) jobs, 2 threads each (64 floats ea).
  {
    int rowId = tid >> 1;        // 0..127
    int e     = rowId >> 1;      // 0..63
    int part  = rowId & 1;       // 0: h[src] -> K 0..127, 1: h[dst] -> K 128..255
    int half  = tid & 1;         // which 64 floats of the 128
    int ge    = base + e;
    bool valid = ge < nEdges;
    long long node = 0;
    if (valid) node = edges[2LL * ge + part];
    if (part == 1 && half == 0) dstLds[e] = valid ? (int)node : 0;
    const float* hp = h + (long)node * HD + half * 64;
    int k0 = part * HD + half * 64;
#pragma unroll
    for (int j = 0; j < 16; ++j) {
      v4f v = {};
      if (valid) v = *(const v4f*)(hp + 4 * j);
      v4bf ph, pl;
      split4(v, &ph, &pl);
      *(v4bf*)&Ahi[e][k0 + 4 * j] = ph;
      *(v4bf*)&Alo[e][k0 + 4 * j] = pl;
    }
  }
  __syncthreads();

  const int wave  = tid >> 5;
  const int lane  = tid & 31;
  const int laneN = lane & 15;
  const int laneH = lane >> 4;   // 0: K-base 0 / rows 0-7, 1: K-base 8 / rows 8-15

#pragma unroll 1
  for (int ntI = 0; ntI < 2; ++ntI) {
    int nt = wave * 2 + ntI;     // 16 output N-tiles over 8 waves
    int o  = nt * 16 + laneN;    // this lane's output column (B layout)
    const __bf16* wrh = Wh + (long)o * HD2;
    const __bf16* wrl = Wl + (long)o * HD2;
    v8f acc[4] = {};
    // Cap unroll to keep live fragments ~130 VGPRs (no scratch spills).
#pragma unroll 2
    for (int kb = 0; kb < 8; ++kb) {   // K = 256 in steps of 32
      int k0 = kb * 32 + laneH * 8;
      int k1 = k0 + 16;
      v16bf bh = load_frag(wrh + k0, wrh + k1);
      v16bf bl = load_frag(wrl + k0, wrl + k1);
#pragma unroll
      for (int mt = 0; mt < 4; ++mt) {
        int r = mt * 16 + laneN;       // A row (edge within tile)
        v16bf ah = load_frag(&Ahi[r][k0], &Ahi[r][k1]);
        v16bf al = load_frag(&Alo[r][k0], &Alo[r][k1]);
        acc[mt] = wmma_split(ah, al, bh, bl, acc[mt]);
      }
    }
    // Scatter-accumulate: C VGPR i holds row M = i + 8*laneH, col N = laneN.
#pragma unroll
    for (int mt = 0; mt < 4; ++mt) {
#pragma unroll
      for (int i = 0; i < 8; ++i) {
        int m  = mt * 16 + laneH * 8 + i;
        int ge = base + m;
        if (ge < nEdges) {
          int d = dstLds[m];
          atomicAdd(&agg[(long)d * HD2 + o], acc[mt][i]);
        }
      }
    }
  }
}

// ---- Node update: gi = (agg + deg*b_msg) @ W_ih^T + b_ih; gh = h @ W_hh^T + b_hh; GRU gate ----
__global__ __launch_bounds__(256, 2) void update_kernel(
    const float* __restrict__ hcur, const float* __restrict__ agg,
    const int* __restrict__ deg,
    const __bf16* __restrict__ Wih_h, const __bf16* __restrict__ Wih_l,
    const __bf16* __restrict__ Whh_h, const __bf16* __restrict__ Whh_l,
    const float* __restrict__ b_msg, const float* __restrict__ b_ih,
    const float* __restrict__ b_hh,
    float* __restrict__ hout, int nNodes)
{
  __shared__ __align__(16) __bf16 Ahi[NTILE][HD2 + 8];
  __shared__ __align__(16) __bf16 Alo[NTILE][HD2 + 8];
  __shared__ __align__(16) __bf16 Hhi[NTILE][HD + 8];
  __shared__ __align__(16) __bf16 Hlo[NTILE][HD + 8];
  __shared__ float GI[NTILE][HD3 + 4];
  __shared__ float GH[NTILE][HD3 + 4];

  const int tid = threadIdx.x;
  const int nb  = blockIdx.x * NTILE;

  // Stage A operands (bf16 hi/lo). agg gets the folded edge bias deg*b_msg.
  {
    int nd = tid >> 4;                 // node within tile
    int n  = nb + nd;
    bool valid = n < nNodes;
    float dg = valid ? (float)deg[n] : 0.0f;
    int kbase = (tid & 15) * 16;
#pragma unroll
    for (int jj = 0; jj < 4; ++jj) {
      int k = kbase + jj * 4;
      v4f v = {};
      if (valid) {
        v = *(const v4f*)(agg + (long)n * HD2 + k);
        v4f bm = *(const v4f*)(b_msg + k);
#pragma unroll
        for (int c = 0; c < 4; ++c) v[c] += dg * bm[c];
      }
      v4bf ph, pl;
      split4(v, &ph, &pl);
      *(v4bf*)&Ahi[nd][k] = ph;
      *(v4bf*)&Alo[nd][k] = pl;
    }
    int hbase = (tid & 15) * 8;
#pragma unroll
    for (int jj = 0; jj < 2; ++jj) {
      int k = hbase + jj * 4;
      v4f v = {};
      if (valid) v = *(const v4f*)(hcur + (long)n * HD + k);
      v4bf ph, pl;
      split4(v, &ph, &pl);
      *(v4bf*)&Hhi[nd][k] = ph;
      *(v4bf*)&Hlo[nd][k] = pl;
    }
  }
  __syncthreads();

  const int wave  = tid >> 5;
  const int lane  = tid & 31;
  const int laneN = lane & 15;
  const int laneH = lane >> 4;

  // 24 N-tiles (384 outputs) for gi and for gh; each wave handles 3 of each.
#pragma unroll 1
  for (int q = 0; q < 3; ++q) {
    int nt = wave * 3 + q;
    int o  = nt * 16 + laneN;
    {
      const __bf16* wrh = Wih_h + (long)o * HD2;
      const __bf16* wrl = Wih_l + (long)o * HD2;
      v8f acc = {};
#pragma unroll 2
      for (int kb = 0; kb < 8; ++kb) {     // K = 256
        int k0 = kb * 32 + laneH * 8, k1 = k0 + 16;
        v16bf bh = load_frag(wrh + k0, wrh + k1);
        v16bf bl = load_frag(wrl + k0, wrl + k1);
        v16bf ah = load_frag(&Ahi[laneN][k0], &Ahi[laneN][k1]);
        v16bf al = load_frag(&Alo[laneN][k0], &Alo[laneN][k1]);
        acc = wmma_split(ah, al, bh, bl, acc);
      }
#pragma unroll
      for (int i = 0; i < 8; ++i)
        GI[laneH * 8 + i][o] = acc[i] + b_ih[o];
    }
    {
      const __bf16* wrh = Whh_h + (long)o * HD;
      const __bf16* wrl = Whh_l + (long)o * HD;
      v8f acc = {};
#pragma unroll 2
      for (int kb = 0; kb < 4; ++kb) {     // K = 128
        int k0 = kb * 32 + laneH * 8, k1 = k0 + 16;
        v16bf bh = load_frag(wrh + k0, wrh + k1);
        v16bf bl = load_frag(wrl + k0, wrl + k1);
        v16bf ah = load_frag(&Hhi[laneN][k0], &Hhi[laneN][k1]);
        v16bf al = load_frag(&Hlo[laneN][k0], &Hlo[laneN][k1]);
        acc = wmma_split(ah, al, bh, bl, acc);
      }
#pragma unroll
      for (int i = 0; i < 8; ++i)
        GH[laneH * 8 + i][o] = acc[i] + b_hh[o];
    }
  }
  __syncthreads();

  // GRU gating: 16 nodes x 128 dims, two 4-wide chunks per thread.
#pragma unroll
  for (int jj = 0; jj < 2; ++jj) {
    int idx = tid * 8 + jj * 4;
    int nd = idx >> 7, dd = idx & 127;
    int n = nb + nd;
    if (n < nNodes) {
      v4f ir = *(const v4f*)&GI[nd][dd];
      v4f iz = *(const v4f*)&GI[nd][HD + dd];
      v4f in = *(const v4f*)&GI[nd][2 * HD + dd];
      v4f hr = *(const v4f*)&GH[nd][dd];
      v4f hz = *(const v4f*)&GH[nd][HD + dd];
      v4f hn = *(const v4f*)&GH[nd][2 * HD + dd];
      v4f hold = *(const v4f*)(hcur + (long)n * HD + dd);
      bool hasmsg = deg[n] > 0;
      v4f out;
#pragma unroll
      for (int c = 0; c < 4; ++c) {
        float r = 1.0f / (1.0f + expf(-(ir[c] + hr[c])));
        float z = 1.0f / (1.0f + expf(-(iz[c] + hz[c])));
        float nn = tanhf(in[c] + r * hn[c]);
        float hnew = (1.0f - z) * nn + z * hold[c];
        out[c] = hasmsg ? hnew : hold[c];
      }
      *(v4f*)(hout + (long)n * HD + dd) = out;
    }
  }
}

static inline size_t alignup(size_t x) { return (x + 255) & ~(size_t)255; }

extern "C" void kernel_launch(void* const* d_in, const int* in_sizes, int n_in,
                              void* d_out, int out_size, void* d_ws, size_t ws_size,
                              hipStream_t stream) {
  (void)n_in; (void)out_size; (void)ws_size;
  const float*     h0    = (const float*)d_in[0];
  const long long* edges = (const long long*)d_in[1];
  const float*     W_msg = (const float*)d_in[2];
  const float*     b_msg = (const float*)d_in[3];
  const float*     W_ih  = (const float*)d_in[4];
  const float*     W_hh  = (const float*)d_in[5];
  const float*     b_ih  = (const float*)d_in[6];
  const float*     b_hh  = (const float*)d_in[7];

  const int  nNodes = in_sizes[0] / HD;           // 50000
  const long nEdges = (long)in_sizes[1] / 2;      // 800000
  const int  T      = in_sizes[2] / (HD2 * HD2);  // 2

  // Workspace carve-up
  char* ws = (char*)d_ws;
  float* agg = (float*)ws;            ws += alignup((size_t)nNodes * HD2 * sizeof(float));
  float* hA  = (float*)ws;            ws += alignup((size_t)nNodes * HD * sizeof(float));
  float* hB  = (float*)ws;            ws += alignup((size_t)nNodes * HD * sizeof(float));
  int*   deg = (int*)ws;              ws += alignup((size_t)nNodes * sizeof(int));
  long nWm = (long)T * HD2 * HD2, nWi = (long)T * HD3 * HD2, nWh = (long)T * HD3 * HD;
  __bf16* Wm_h = (__bf16*)ws;         ws += alignup(nWm * sizeof(__bf16));
  __bf16* Wm_l = (__bf16*)ws;         ws += alignup(nWm * sizeof(__bf16));
  __bf16* Wi_h = (__bf16*)ws;         ws += alignup(nWi * sizeof(__bf16));
  __bf16* Wi_l = (__bf16*)ws;         ws += alignup(nWi * sizeof(__bf16));
  __bf16* Wh_h = (__bf16*)ws;         ws += alignup(nWh * sizeof(__bf16));
  __bf16* Wh_l = (__bf16*)ws;         ws += alignup(nWh * sizeof(__bf16));

  // Degree (constant across timesteps)
  zero_i32_kernel<<<256, 256, 0, stream>>>(deg, nNodes);
  deg_kernel<<<(int)((nEdges + 255) / 256), 256, 0, stream>>>(edges, deg, nEdges);

  // bf16 hi/lo weight splits (done once per launch)
  cvt_kernel<<<512, 256, 0, stream>>>(W_msg, Wm_h, Wm_l, nWm);
  cvt_kernel<<<512, 256, 0, stream>>>(W_ih,  Wi_h, Wi_l, nWi);
  cvt_kernel<<<512, 256, 0, stream>>>(W_hh,  Wh_h, Wh_l, nWh);

  const float* hc = h0;
  for (int t = 0; t < T; ++t) {
    zero_f32_kernel<<<4096, 256, 0, stream>>>(agg, (long)nNodes * HD2);
    msg_kernel<<<(int)((nEdges + ETILE - 1) / ETILE), 256, 0, stream>>>(
        hc, edges, Wm_h + (long)t * HD2 * HD2, Wm_l + (long)t * HD2 * HD2, agg, (int)nEdges);
    float* ho = (t == T - 1) ? (float*)d_out : ((t & 1) ? hB : hA);
    update_kernel<<<(nNodes + NTILE - 1) / NTILE, 256, 0, stream>>>(
        hc, agg, deg,
        Wi_h + (long)t * HD3 * HD2, Wi_l + (long)t * HD3 * HD2,
        Wh_h + (long)t * HD3 * HD,  Wh_l + (long)t * HD3 * HD,
        b_msg + (long)t * HD2, b_ih + (long)t * HD3, b_hh + (long)t * HD3,
        ho, nNodes);
    hc = ho;
  }
}
